// MultiHeadAttention_2645699854825
// MI455X (gfx1250) — compile-verified, matrix operand order checked by reading, not run on hardware
//
#include <hip/hip_runtime.h>
#include <stdint.h>

// ---------------------------------------------------------------------------
// MHA forward for MI455X (gfx1250): bf16 WMMA (16x16x32, fp32 accum) GEMMs +
// flash-attention with online softmax. Wave32 layout math throughout.
// ---------------------------------------------------------------------------

typedef __bf16 bf16_t;
typedef __attribute__((ext_vector_type(16))) __bf16 v16bf;
typedef __attribute__((ext_vector_type(8)))  __bf16 v8bf;
typedef __attribute__((ext_vector_type(8)))  float  v8f;

#define D_MODEL 1024
#define N_HEADS 16
#define D_K     64
#define BATCH   2
#define SEQ     2048
#define NTOK    (BATCH * SEQ)          // 4096
#define NSLAB   (D_MODEL / 32)         // 32 K-slabs per GEMM

// Native bf16 truncation (single v_cvt on gfx1250)
__device__ __forceinline__ bf16_t f2bf(float f) { return (bf16_t)f; }

// Load a 16-element bf16 fragment per lane. Caller passes (base + half*8).
// ISA 16-bit A/B layout: element e -> K = (e<8 ? e : e+8) + half*8, so the
// fragment is two contiguous 8-element (16-byte) chunks at +0 and +16.
__device__ __forceinline__ v16bf load_frag(const bf16_t* __restrict__ p) {
  v8bf lo = *(const v8bf*)(p);
  v8bf hi = *(const v8bf*)(p + 16);
  v16bf r;
#pragma unroll
  for (int i = 0; i < 8; ++i) { r[i] = lo[i]; r[i + 8] = hi[i]; }
  return r;
}

__device__ __forceinline__ v8f wmma_bf16(v16bf a, v16bf b, v8f c) {
  // D = A x B + C, 16x16x32 bf16 -> f32   (v_wmma_f32_16x16x32_bf16)
  return __builtin_amdgcn_wmma_f32_16x16x32_bf16(
      /*neg_a=*/false, a, /*neg_b=*/false, b,
      /*c_mod=*/(short)0, c, /*reuse_a=*/false, /*reuse_b=*/false);
}

// Fragment buffer for one 32-wide K-slab of a 32(M) x 64(N) tile.
struct Frags { v16bf a[2]; v16bf b[4]; };

__device__ __forceinline__ void load_slab(Frags& f,
                                          const bf16_t* const* pa,
                                          const bf16_t* const* pb, int kk) {
#pragma unroll
  for (int ms = 0; ms < 2; ++ms) f.a[ms] = load_frag(pa[ms] + kk);
#pragma unroll
  for (int t = 0; t < 4; ++t)    f.b[t]  = load_frag(pb[t] + kk);
}

__device__ __forceinline__ void mma_slab(v8f (&acc)[2][4], const Frags& f) {
#pragma unroll
  for (int t = 0; t < 4; ++t)
#pragma unroll
    for (int ms = 0; ms < 2; ++ms)
      acc[ms][t] = wmma_bf16(f.a[ms], f.b[t], acc[ms][t]);
}

// ---------------------------------------------------------------------------
// Precision conversion / weight transpose (one-shot, bandwidth-trivial)
// ---------------------------------------------------------------------------
__global__ void cvt_bf16_kernel(const float* __restrict__ src,
                                bf16_t* __restrict__ dst, int n) {
  int i = blockIdx.x * blockDim.x + threadIdx.x;
  if (i < n) dst[i] = f2bf(src[i]);
}

// src: [K,N] fp32 row-major  ->  dst: [N,K] bf16 row-major
__global__ void transpose_bf16_kernel(const float* __restrict__ src,
                                      bf16_t* __restrict__ dst, int K, int N) {
  int i = blockIdx.x * blockDim.x + threadIdx.x;
  if (i < K * N) {
    int n = i / K, k = i - n * K;
    dst[i] = f2bf(src[k * N + n]);
  }
}

// ---------------------------------------------------------------------------
// GEMM 1: Y[4096,3072] = Xb[4096,1024] @ Wqkv + bqkv, epilogue scatters into
// Q [b,h,s,d], K [b,h,s,d] (bf16) and V transposed [b,h,d,s] (bf16) so the
// attention kernel gets contiguous WMMA fragments for all three.
// One wave computes a 32(M) x 64(N) tile; explicit ping-pong double buffer
// over K (no register copies).  waves = 128 * 48 = 6144.
// amdgpu_waves_per_eu(2): allow ~half the VGPR file per wave so the
// double-buffered tile (acc 64 + 2x48 frag regs + addrs) allocates cleanly.
// ---------------------------------------------------------------------------
__global__ void __launch_bounds__(256)
__attribute__((amdgpu_waves_per_eu(2)))
gemm_qkv_kernel(const bf16_t* __restrict__ A,    // [4096,1024]
                const bf16_t* __restrict__ Bt,   // [3072,1024]
                const float*  __restrict__ bias, // [3072]
                bf16_t* __restrict__ Qb,
                bf16_t* __restrict__ Kb,
                bf16_t* __restrict__ Vt) {
  const int wid   = (blockIdx.x * blockDim.x + threadIdx.x) >> 5;
  const int lane  = threadIdx.x & 31;
  const int colL  = lane & 15;
  const int hf    = lane >> 4;
  const int mtile = wid & 127;          // 128 M tiles of 32 rows
  const int ntile = wid >> 7;           // 48 N tiles of 64 cols
  const int m0 = mtile * 32, n0 = ntile * 64;

  const bf16_t* pa[2];
  const bf16_t* pb[4];
#pragma unroll
  for (int ms = 0; ms < 2; ++ms)
    pa[ms] = A + (size_t)(m0 + ms * 16 + colL) * D_MODEL + hf * 8;
#pragma unroll
  for (int t = 0; t < 4; ++t)
    pb[t] = Bt + (size_t)(n0 + t * 16 + colL) * D_MODEL + hf * 8;

  v8f acc[2][4] = {};
  Frags f0, f1;
  load_slab(f0, pa, pb, 0);
  // slabs 1..30 in ping-pong pairs; loads land directly in WMMA source regs
  for (int i = 0; i < (NSLAB - 2) / 2; ++i) {
    load_slab(f1, pa, pb, (2 * i + 1) * 32);
    mma_slab(acc, f0);
    load_slab(f0, pa, pb, (2 * i + 2) * 32);
    mma_slab(acc, f1);
  }
  load_slab(f1, pa, pb, (NSLAB - 1) * 32);
  mma_slab(acc, f0);
  mma_slab(acc, f1);

  // n0 is a multiple of 64 -> the whole 64-wide tile is one (q|k|v, head).
  const int which = n0 >> 10;            // 0=q 1=k 2=v
  const int head  = (n0 & 1023) >> 6;

#pragma unroll
  for (int t = 0; t < 4; ++t) {
    const int c  = n0 + t * 16 + colL;
    const float bv = bias[c];
    const int d  = t * 16 + colL;        // 0..63 within head
#pragma unroll
    for (int ms = 0; ms < 2; ++ms) {
#pragma unroll
      for (int r = 0; r < 8; ++r) {
        const int tok = m0 + ms * 16 + r + hf * 8;
        const int b_  = tok >> 11;       // / SEQ
        const int s_  = tok & (SEQ - 1);
        bf16_t o = f2bf(acc[ms][t][r] + bv);
        if (which == 0)
          Qb[(((size_t)(b_ * N_HEADS + head) * SEQ) + s_) * D_K + d] = o;
        else if (which == 1)
          Kb[(((size_t)(b_ * N_HEADS + head) * SEQ) + s_) * D_K + d] = o;
        else
          Vt[(((size_t)(b_ * N_HEADS + head) * D_K) + d) * SEQ + s_] = o;
      }
    }
  }
}

// ---------------------------------------------------------------------------
// Flash attention: one wave (32 threads) per (b,h, 16-row q tile).
// Streams causal-live 64-key blocks: 8 WMMAs for scores, online softmax with
// __shfl_xor row reductions (amortized over 64 keys), P through LDS (layout
// swap), 8 WMMAs for P*V.  V^T layout gives contiguous B fragments.
// ---------------------------------------------------------------------------
#define P_STRIDE 72   // padded row stride (elements): 144B, 16B-aligned, bank-staggered

__global__ void __launch_bounds__(32)
__attribute__((amdgpu_waves_per_eu(2)))
attn_kernel(const bf16_t* __restrict__ Qb,
            const bf16_t* __restrict__ Kb,
            const bf16_t* __restrict__ Vt,
            bf16_t* __restrict__ Ctx) {
  __shared__ __align__(16) bf16_t pT[16 * P_STRIDE];

  const int lane = threadIdx.x & 31;
  const int colL = lane & 15;
  const int hf   = lane >> 4;
  const int qt   = blockIdx.x & 127;           // 128 q tiles
  const int bh   = blockIdx.x >> 7;            // 0..31 = b*16+h
  const int q0   = qt * 16;

  const bf16_t* Qh = Qb + (size_t)bh * SEQ * D_K;
  const bf16_t* Kh = Kb + (size_t)bh * SEQ * D_K;
  const bf16_t* Vh = Vt + (size_t)bh * D_K * SEQ;

  // Q A-fragments for both dk chunks (lane = M row = colL)
  const v16bf aq0 = load_frag(Qh + (size_t)(q0 + colL) * D_K + hf * 8);
  const v16bf aq1 = load_frag(Qh + (size_t)(q0 + colL) * D_K + 32 + hf * 8);

  v8f o[4] = {};
  float m[8], l[8];
#pragma unroll
  for (int r = 0; r < 8; ++r) { m[r] = -__builtin_inff(); l[r] = 0.f; }

  const float scale = 0.125f;                  // 1/sqrt(64)
  const int nblk = ((q0 + 15) >> 6) + 1;       // causal-live 64-key blocks

  for (int j = 0; j < nblk; ++j) {
    const int k0 = j * 64;

    // ---- scores: S[16 x 64] = Q(16x64) . K^T, four 16-col subtiles ----
    v8f s[4] = {};
#pragma unroll
    for (int st = 0; st < 4; ++st) {
      v16bf kbA = load_frag(Kh + (size_t)(k0 + st * 16 + colL) * D_K + hf * 8);
      v16bf kbB = load_frag(Kh + (size_t)(k0 + st * 16 + colL) * D_K + 32 + hf * 8);
      s[st] = wmma_bf16(aq0, kbA, s[st]);
      s[st] = wmma_bf16(aq1, kbB, s[st]);
    }

    // ---- scale + causal mask + online softmax (row = q0 + r + hf*8) ----
#pragma unroll
    for (int r = 0; r < 8; ++r) {
      const int row = q0 + r + hf * 8;
      float x[4];
      float mx = -__builtin_inff();
#pragma unroll
      for (int st = 0; st < 4; ++st) {
        const int c = k0 + st * 16 + colL;
        x[st] = (c <= row) ? s[st][r] * scale : -__builtin_inff();
        mx = fmaxf(mx, x[st]);
      }
      mx = fmaxf(mx, __shfl_xor(mx, 1));
      mx = fmaxf(mx, __shfl_xor(mx, 2));
      mx = fmaxf(mx, __shfl_xor(mx, 4));
      mx = fmaxf(mx, __shfl_xor(mx, 8));
      const float mnew = fmaxf(m[r], mx);
      const float corr = __expf(m[r] - mnew);
      float p[4], ps = 0.f;
#pragma unroll
      for (int st = 0; st < 4; ++st) { p[st] = __expf(x[st] - mnew); ps += p[st]; }
      ps += __shfl_xor(ps, 1);
      ps += __shfl_xor(ps, 2);
      ps += __shfl_xor(ps, 4);
      ps += __shfl_xor(ps, 8);
      l[r] = l[r] * corr + ps;
      m[r] = mnew;
#pragma unroll
      for (int t = 0; t < 4; ++t) o[t][r] = o[t][r] * corr;

      const int lrow = r + hf * 8;               // local row 0..15
#pragma unroll
      for (int st = 0; st < 4; ++st)
        pT[lrow * P_STRIDE + st * 16 + colL] = f2bf(p[st]);
    }

    // LDS in-order per wave; make stores visible to the fragment reloads.
    asm volatile("s_wait_dscnt 0" ::: "memory");

    // ---- P as A-fragments: lane = row = colL, elements = keys ----
    v16bf ap0 = load_frag(pT + colL * P_STRIDE + hf * 8);        // keys 0..31
    v16bf ap1 = load_frag(pT + colL * P_STRIDE + 32 + hf * 8);   // keys 32..63

    // ---- O += P(16x64) . V(64x64): V^T layout gives contiguous B frags ----
#pragma unroll
    for (int t = 0; t < 4; ++t) {
      v16bf vb0 = load_frag(Vh + (size_t)(t * 16 + colL) * SEQ + k0 + hf * 8);
      o[t] = wmma_bf16(ap0, vb0, o[t]);
      v16bf vb1 = load_frag(Vh + (size_t)(t * 16 + colL) * SEQ + k0 + 32 + hf * 8);
      o[t] = wmma_bf16(ap1, vb1, o[t]);
    }
  }

  // ---- normalize and store context [token, h*64+d] as bf16 ----
  const int b_ = bh >> 4, h_ = bh & 15;
#pragma unroll
  for (int r = 0; r < 8; ++r) {
    const float inv = 1.f / l[r];
    const int tok = b_ * SEQ + q0 + r + hf * 8;
    bf16_t* crow = Ctx + (size_t)tok * D_MODEL + h_ * D_K;
#pragma unroll
    for (int t = 0; t < 4; ++t)
      crow[t * 16 + colL] = f2bf(o[t][r] * inv);
  }
}

// ---------------------------------------------------------------------------
// GEMM 2: out[4096,1024] = Ctx[4096,1024](bf16) @ Wout + bout  (fp32 out)
// One wave computes a 32 x 64 tile; ping-pong double buffer. waves = 128*16.
// ---------------------------------------------------------------------------
__global__ void __launch_bounds__(256)
__attribute__((amdgpu_waves_per_eu(2)))
gemm_out_kernel(const bf16_t* __restrict__ A,    // [4096,1024]
                const bf16_t* __restrict__ Bt,   // [1024,1024]
                const float*  __restrict__ bias, // [1024]
                float* __restrict__ out) {
  const int wid   = (blockIdx.x * blockDim.x + threadIdx.x) >> 5;
  const int lane  = threadIdx.x & 31;
  const int colL  = lane & 15;
  const int hf    = lane >> 4;
  const int mtile = wid & 127;
  const int ntile = wid >> 7;            // 0..15
  const int m0 = mtile * 32, n0 = ntile * 64;

  const bf16_t* pa[2];
  const bf16_t* pb[4];
#pragma unroll
  for (int ms = 0; ms < 2; ++ms)
    pa[ms] = A + (size_t)(m0 + ms * 16 + colL) * D_MODEL + hf * 8;
#pragma unroll
  for (int t = 0; t < 4; ++t)
    pb[t] = Bt + (size_t)(n0 + t * 16 + colL) * D_MODEL + hf * 8;

  v8f acc[2][4] = {};
  Frags f0, f1;
  load_slab(f0, pa, pb, 0);
  for (int i = 0; i < (NSLAB - 2) / 2; ++i) {
    load_slab(f1, pa, pb, (2 * i + 1) * 32);
    mma_slab(acc, f0);
    load_slab(f0, pa, pb, (2 * i + 2) * 32);
    mma_slab(acc, f1);
  }
  load_slab(f1, pa, pb, (NSLAB - 1) * 32);
  mma_slab(acc, f0);
  mma_slab(acc, f1);

#pragma unroll
  for (int t = 0; t < 4; ++t) {
    const int c = n0 + t * 16 + colL;
    const float bv = bias[c];
#pragma unroll
    for (int ms = 0; ms < 2; ++ms) {
#pragma unroll
      for (int r = 0; r < 8; ++r) {
        const int tok = m0 + ms * 16 + r + hf * 8;
        out[(size_t)tok * D_MODEL + c] = acc[ms][t][r] + bv;
      }
    }
  }
}

// ---------------------------------------------------------------------------
// Host-side launch (graph-capture safe: stream-only, workspace scratch)
// ---------------------------------------------------------------------------
extern "C" void kernel_launch(void* const* d_in, const int* in_sizes, int n_in,
                              void* d_out, int out_size, void* d_ws, size_t ws_size,
                              hipStream_t stream) {
  const float* x    = (const float*)d_in[0];
  // d_in[1] = mask (bool tril) — causal mask applied analytically in-kernel
  const float* Wqkv = (const float*)d_in[2];
  const float* bqkv = (const float*)d_in[3];
  const float* Wout = (const float*)d_in[4];
  const float* bout = (const float*)d_in[5];
  float* out = (float*)d_out;

  char* ws = (char*)d_ws;
  bf16_t* xb    = (bf16_t*)(ws + (size_t)( 0u << 20));  //  8 MB  x in bf16
  bf16_t* WqkvT = (bf16_t*)(ws + (size_t)( 8u << 20));  //  6 MB  Wqkv^T bf16
  bf16_t* WoutT = (bf16_t*)(ws + (size_t)(14u << 20));  //  2 MB  Wout^T bf16
  bf16_t* Qb    = (bf16_t*)(ws + (size_t)(16u << 20));  //  8 MB  [b,h,s,d]
  bf16_t* Kb    = (bf16_t*)(ws + (size_t)(24u << 20));  //  8 MB  [b,h,s,d]
  bf16_t* Vt    = (bf16_t*)(ws + (size_t)(32u << 20));  //  8 MB  [b,h,d,s]
  bf16_t* Ctx   = (bf16_t*)(ws + (size_t)(40u << 20));  //  8 MB  [tok, dmodel]

  cvt_bf16_kernel<<<(NTOK * D_MODEL) / 256, 256, 0, stream>>>(x, xb, NTOK * D_MODEL);
  transpose_bf16_kernel<<<(D_MODEL * 3 * D_MODEL) / 256, 256, 0, stream>>>(
      Wqkv, WqkvT, D_MODEL, 3 * D_MODEL);
  transpose_bf16_kernel<<<(D_MODEL * D_MODEL) / 256, 256, 0, stream>>>(
      Wout, WoutT, D_MODEL, D_MODEL);

  // 6144 waves, 8 waves / 256-thread block
  gemm_qkv_kernel<<<768, 256, 0, stream>>>(xb, WqkvT, bqkv, Qb, Kb, Vt);

  // one wave32 per (b,h,q-tile): 2*16*128 = 4096 workgroups
  attn_kernel<<<BATCH * N_HEADS * (SEQ / 16), 32, 0, stream>>>(Qb, Kb, Vt, Ctx);

  // 2048 waves, 8 waves / block
  gemm_out_kernel<<<256, 256, 0, stream>>>(Ctx, WoutT, bout, out);
}
